// WindowAttention_61289183314175
// MI455X (gfx1250) — matrix operand order repaired and checked by reading, not compiled, not run on hardware
//
#include <hip/hip_runtime.h>

#define NTOK 49
#define NP   64
#define DIM  384
#define HEADS 12
#define HD   32
#define SCALEF 0.1767766952966369f   // 32^-0.5

typedef __attribute__((ext_vector_type(16))) __bf16 bf16x16;
typedef __attribute__((ext_vector_type(8)))  float  f32x8;

// packed fragment sizes (elements)
#define JQ (72*12*512)   // w_qkv B-fragments
#define JP (24*12*512)   // w_proj B-fragments
#define JB (HEADS*4*4*32*8) // bias in C-fragment order [h][mt][nt][lane][r]

// --- fragment index helpers (bf16 WMMA 16x16x32, wave32, contiguous k order used by both A and B) ---
// A (16x32, MxK): lanes 0-15 hold K 0..15; lanes 16-31 hold K 16..31
__device__ __forceinline__ int afrag_idx(int mr, int kk) {
  int ln = mr + 16 * (kk >> 4);
  int j  = (kk & 15) >> 1;
  return ln * 16 + j * 2 + (kk & 1);
}
// B (32x16, KxN): lanes 0-15 hold K 0..15 for col n; lanes 16-31 hold K 16..31
__device__ __forceinline__ int bfrag_idx(int kk, int n) {
  int ln = n + 16 * (kk >> 4);
  int j  = (kk & 15) >> 1;
  return ln * 16 + j * 2 + (kk & 1);
}

// ---------------- prologue: pack weights (f32 -> bf16 B-frags) + C-layout bias ----------------
__global__ __launch_bounds__(256) void pack_params(
    const float* __restrict__ w_qkv, const float* __restrict__ w_proj,
    const float* __restrict__ bias_table, const int* __restrict__ rel_idx,
    __bf16* __restrict__ wq, __bf16* __restrict__ wp, float* __restrict__ bias_pad)
{
  int i = blockIdx.x * 256 + threadIdx.x;
  if (i < JQ) {
    int slot = i & 1, j = (i >> 1) & 7, lane = (i >> 4) & 31;
    int grp = i >> 9;
    int ks = grp % 12, nt = grp / 12;
    int kk = 2 * j + slot + 16 * (lane >> 4);
    int col = nt * 16 + (lane & 15);
    wq[i] = (__bf16)w_qkv[(ks * 32 + kk) * 1152 + col];
  } else if (i < JQ + JP) {
    int t = i - JQ;
    int slot = t & 1, j = (t >> 1) & 7, lane = (t >> 4) & 31;
    int grp = t >> 9;
    int ks = grp % 12, nt = grp / 12;
    int kk = 2 * j + slot + 16 * (lane >> 4);
    int col = nt * 16 + (lane & 15);
    wp[t] = (__bf16)w_proj[(ks * 32 + kk) * 384 + col];
  } else if (i < JQ + JP + JB) {
    // bias in WMMA C/D fragment order: [h][mt][nt][lane][r]
    int t = i - JQ - JP;
    int r = t & 7, lane = (t >> 3) & 31, nt = (t >> 8) & 3, mt = (t >> 10) & 3, h = t >> 12;
    int row = mt * 16 + r + 8 * (lane >> 4);
    int col = nt * 16 + (lane & 15);
    float v = 0.f;
    if (row < NTOK && col < NTOK) v = bias_table[rel_idx[row * NTOK + col] * HEADS + h];
    bias_pad[t] = v;
  }
}

// ---------------- fused window attention: one workgroup per window ----------------
__global__ __launch_bounds__(384) void window_attn(
    const float* __restrict__ x,
    const __bf16* __restrict__ wq,
    const __bf16* __restrict__ wp,
    const float* __restrict__ bias_pad,
    const float* __restrict__ b_proj,
    float* __restrict__ out)
{
  // x / attention-output as A-fragments: [4 mtiles][12 ksteps][32 lanes][16 bf16]
  __shared__ __attribute__((aligned(32))) __bf16 xs[4 * 12 * 512];
  __shared__ __attribute__((aligned(32))) __bf16 outp[4 * 12 * 512];
  // per-wave scratch: Q A-frags [0,2048), K^T B-frags [2048,4096), V B-frags [4096,6144)
  // P A-frags overlay [0,4096) after S is computed
  __shared__ __attribute__((aligned(32))) __bf16 wsc[12][6144];

  const int tid  = threadIdx.x;
  const int lane = tid & 31;
  const int w    = tid >> 5;          // wave id == head id (12 waves)
  const int colb = lane & 15;
  const int rowoff = (lane >> 4) * 8; // C/D layout: lanes 16-31 hold rows +8
  const size_t b = blockIdx.x;
  const float* xb = x + b * (size_t)(NTOK * DIM);

  const f32x8 zero8 = {0.f,0.f,0.f,0.f,0.f,0.f,0.f,0.f};

  // ---- load window (row m = iteration, col k = tid since blockDim==DIM) ----
  {
    const int kofs = (tid >> 5) * 512 + afrag_idx(0, tid & 31); // m-invariant part
#pragma unroll 4
    for (int m = 0; m < NP; ++m) {
      float v = (m < NTOK) ? xb[m * DIM + tid] : 0.f;
      xs[(m >> 4) * (12 * 512) + (m & 15) * 16 + kofs] = (__bf16)v;
    }
  }
  __syncthreads();

  __bf16* myS = &wsc[w][0];     // base of this wave's scratch
  __bf16* myV = &wsc[w][4096];
  __bf16* myP = &wsc[w][0];

  const __bf16* xbase = xs + lane * 16;

  // ---- QKV GEMM for this head: 6 output ntiles (2 each of q/k/v), K=384 ----
#pragma unroll 1
  for (int d = 0; d < 6; ++d) {
    asm volatile("" ::: "memory");            // block LICM of LDS loads across d
    const int seg = d >> 1;                   // 0:q 1:k 2:v
    const int ntd = d & 1;                    // dim tile within head
    const int ntg = seg * 24 + w * 2 + ntd;   // global ntile in 0..71
    const __bf16* wbase = wq + (size_t)(ntg * 12) * 512 + lane * 16;
    f32x8 acc[4];
    bf16x16 bA = *(const bf16x16*)(wbase);    // 2-stage pipelined B fragments
    // ---- peeled ks=0,1: accumulate from inline C=0 (no acc zero-init movs) ----
    {
      bf16x16 bB = *(const bf16x16*)(wbase + 512);
      bf16x16 af[4];
#pragma unroll
      for (int mt = 0; mt < 4; ++mt)
        af[mt] = *(const bf16x16*)(xbase + (mt * 12 + 0) * 512);
      __builtin_amdgcn_sched_barrier(0);
#pragma unroll
      for (int mt = 0; mt < 4; ++mt)
        acc[mt] = __builtin_amdgcn_wmma_f32_16x16x32_bf16(
            false, af[mt], false, bA, (short)0, zero8, false, false);
      __builtin_amdgcn_sched_barrier(0);
      bA = *(const bf16x16*)(wbase + 2 * 512);
      bf16x16 ag[4];
#pragma unroll
      for (int mt = 0; mt < 4; ++mt)
        ag[mt] = *(const bf16x16*)(xbase + (mt * 12 + 1) * 512);
      __builtin_amdgcn_sched_barrier(0);
#pragma unroll
      for (int mt = 0; mt < 4; ++mt)
        acc[mt] = __builtin_amdgcn_wmma_f32_16x16x32_bf16(
            false, ag[mt], false, bB, (short)0, acc[mt], false, false);
    }
#pragma unroll 1
    for (int ks = 2; ks < 12; ks += 2) {
      bf16x16 bB = *(const bf16x16*)(wbase + (size_t)(ks + 1) * 512);
      bf16x16 af[4];
#pragma unroll
      for (int mt = 0; mt < 4; ++mt)
        af[mt] = *(const bf16x16*)(xbase + (mt * 12 + ks) * 512);
      __builtin_amdgcn_sched_barrier(0);      // group loads | 4 back-to-back WMMAs
#pragma unroll
      for (int mt = 0; mt < 4; ++mt)
        acc[mt] = __builtin_amdgcn_wmma_f32_16x16x32_bf16(
            false, af[mt], false, bA, (short)0, acc[mt], false, false);
      __builtin_amdgcn_sched_barrier(0);
      if (ks < 10) bA = *(const bf16x16*)(wbase + (size_t)(ks + 2) * 512);
      bf16x16 ag[4];
#pragma unroll
      for (int mt = 0; mt < 4; ++mt)
        ag[mt] = *(const bf16x16*)(xbase + (mt * 12 + ks + 1) * 512);
      __builtin_amdgcn_sched_barrier(0);
#pragma unroll
      for (int mt = 0; mt < 4; ++mt)
        acc[mt] = __builtin_amdgcn_wmma_f32_16x16x32_bf16(
            false, ag[mt], false, bB, (short)0, acc[mt], false, false);
    }
    // branchless repack into per-wave scratch in the layout the next GEMM wants
#pragma unroll
    for (int mt = 0; mt < 4; ++mt) {
#pragma unroll
      for (int r = 0; r < 8; ++r) {
        __bf16 bv = (__bf16)acc[mt][r];
        int ri = r + rowoff;              // token row within tile
        int dc = ntd * 16 + colb;         // head-dim col 0..31
        int tok = mt * 16 + ri;
        int iq = mt * 512 + afrag_idx(ri, dc);                       // Q as A-frag
        int ik = 2048 + mt * 512 + bfrag_idx(dc, ri);                // K^T as B-frag
        int iv = 4096 + ((tok >> 5) * 2 + (dc >> 4)) * 512
                      + bfrag_idx(tok & 31, dc & 15);                // V as B-frag
        int idx = (seg == 0) ? iq : ((seg == 1) ? ik : iv);
        myS[idx] = bv;
      }
    }
  }

  // ---- S = Q K^T, P = (S*scale + bias)^2, row sums (pad rows/cols are exactly 0) ----
  bf16x16 qf[4], kf[4];
#pragma unroll
  for (int mt = 0; mt < 4; ++mt)
    qf[mt] = *(const bf16x16*)(myS + mt * 512 + lane * 16);          // Q
#pragma unroll
  for (int nt = 0; nt < 4; ++nt)
    kf[nt] = *(const bf16x16*)(myS + 2048 + nt * 512 + lane * 16);   // K^T

  const float* bh = bias_pad + w * 4096 + lane * 8;  // C-frag order: [mt][nt][lane][r]
  float inv[4][8];
#pragma unroll
  for (int mt = 0; mt < 4; ++mt) {
    f32x8 sr[4];
#pragma unroll
    for (int nt = 0; nt < 4; ++nt)
      sr[nt] = __builtin_amdgcn_wmma_f32_16x16x32_bf16(
          false, qf[mt], false, kf[nt], (short)0, zero8, false, false);
    float part[8];
#pragma unroll
    for (int r = 0; r < 8; ++r) part[r] = 0.f;
#pragma unroll
    for (int nt = 0; nt < 4; ++nt) {
      f32x8 bias8 = *(const f32x8*)(bh + ((mt * 4 + nt) << 8));  // 32B coalesced
#pragma unroll
      for (int r = 0; r < 8; ++r) {
        int ri  = r + rowoff;
        int col = nt * 16 + colb;
        float a = sr[nt][r] * SCALEF + bias8[r];
        float p = a * a;
        part[r] += p;
        // P as A-frag (K-dim = key token), overlays dead Q/K scratch
        myP[(mt * 2 + (col >> 5)) * 512 + afrag_idx(ri, col & 31)] = (__bf16)p;
      }
    }
#pragma unroll
    for (int r = 0; r < 8; ++r) {        // reduce across the 16 lanes holding one row
      float v = part[r];
      v += __shfl_xor(v, 1);
      v += __shfl_xor(v, 2);
      v += __shfl_xor(v, 4);
      v += __shfl_xor(v, 8);
      inv[mt][r] = 1.f / (v + 1e-6f);
    }
  }

  // ---- O = P V, then normalize by row sum (normalization commutes with P·V) ----
  f32x8 o[4][2];
#pragma unroll
  for (int kst = 0; kst < 2; ++kst) {     // kst==0 uses inline C=0 (folds statically)
    bf16x16 pf[4];
#pragma unroll
    for (int mt = 0; mt < 4; ++mt)
      pf[mt] = *(const bf16x16*)(myP + (mt * 2 + kst) * 512 + lane * 16);
#pragma unroll
    for (int ntd = 0; ntd < 2; ++ntd) {
      bf16x16 vf = *(const bf16x16*)(myV + (kst * 2 + ntd) * 512 + lane * 16);
#pragma unroll
      for (int mt = 0; mt < 4; ++mt)
        o[mt][ntd] = __builtin_amdgcn_wmma_f32_16x16x32_bf16(
            false, pf[mt], false, vf, (short)0,
            (kst == 0) ? zero8 : o[mt][ntd], false, false);
    }
  }
#pragma unroll
  for (int mt = 0; mt < 4; ++mt)
#pragma unroll
    for (int ntd = 0; ntd < 2; ++ntd)
#pragma unroll
      for (int r = 0; r < 8; ++r) {
        float v = o[mt][ntd][r] * inv[mt][r];
        int gcol = w * 32 + ntd * 16 + colb;   // head's slice of the 384 dims
        outp[(mt * 12 + (gcol >> 5)) * 512 + afrag_idx(r + rowoff, gcol & 31)] = (__bf16)v;
      }
  __syncthreads();

  // ---- projection GEMM + bias: 24 ntiles over 12 waves ----
  float* ob = out + b * (size_t)(NTOK * DIM);
  const __bf16* obase = outp + lane * 16;
#pragma unroll 1
  for (int dn = 0; dn < 2; ++dn) {
    asm volatile("" ::: "memory");            // block LICM of LDS loads across dn
    int nt = w * 2 + dn;
    const __bf16* pbase = wp + (size_t)(nt * 12) * 512 + lane * 16;
    f32x8 acc[4];
    bf16x16 bA = *(const bf16x16*)(pbase);
    // ---- peeled ks=0,1 with inline C=0 ----
    {
      bf16x16 bB = *(const bf16x16*)(pbase + 512);
      bf16x16 af[4];
#pragma unroll
      for (int mt = 0; mt < 4; ++mt)
        af[mt] = *(const bf16x16*)(obase + (mt * 12 + 0) * 512);
      __builtin_amdgcn_sched_barrier(0);
#pragma unroll
      for (int mt = 0; mt < 4; ++mt)
        acc[mt] = __builtin_amdgcn_wmma_f32_16x16x32_bf16(
            false, af[mt], false, bA, (short)0, zero8, false, false);
      __builtin_amdgcn_sched_barrier(0);
      bA = *(const bf16x16*)(pbase + 2 * 512);
      bf16x16 ag[4];
#pragma unroll
      for (int mt = 0; mt < 4; ++mt)
        ag[mt] = *(const bf16x16*)(obase + (mt * 12 + 1) * 512);
      __builtin_amdgcn_sched_barrier(0);
#pragma unroll
      for (int mt = 0; mt < 4; ++mt)
        acc[mt] = __builtin_amdgcn_wmma_f32_16x16x32_bf16(
            false, ag[mt], false, bB, (short)0, acc[mt], false, false);
    }
#pragma unroll 1
    for (int ks = 2; ks < 12; ks += 2) {
      bf16x16 bB = *(const bf16x16*)(pbase + (size_t)(ks + 1) * 512);
      bf16x16 af[4];
#pragma unroll
      for (int mt = 0; mt < 4; ++mt)
        af[mt] = *(const bf16x16*)(obase + (mt * 12 + ks) * 512);
      __builtin_amdgcn_sched_barrier(0);
#pragma unroll
      for (int mt = 0; mt < 4; ++mt)
        acc[mt] = __builtin_amdgcn_wmma_f32_16x16x32_bf16(
            false, af[mt], false, bA, (short)0, acc[mt], false, false);
      __builtin_amdgcn_sched_barrier(0);
      if (ks < 10) bA = *(const bf16x16*)(pbase + (size_t)(ks + 2) * 512);
      bf16x16 ag[4];
#pragma unroll
      for (int mt = 0; mt < 4; ++mt)
        ag[mt] = *(const bf16x16*)(obase + (mt * 12 + ks + 1) * 512);
      __builtin_amdgcn_sched_barrier(0);
#pragma unroll
      for (int mt = 0; mt < 4; ++mt)
        acc[mt] = __builtin_amdgcn_wmma_f32_16x16x32_bf16(
            false, ag[mt], false, bB, (short)0, acc[mt], false, false);
    }
    float bp = b_proj[nt * 16 + colb];
    float* obc = ob + nt * 16 + colb;
    // rows 0..47 (mtiles 0..2) are always in range
#pragma unroll
    for (int mt = 0; mt < 3; ++mt)
#pragma unroll
      for (int r = 0; r < 8; ++r)
        obc[(mt * 16 + r + rowoff) * DIM] = acc[mt][r] + bp;
    // mtile 3: only row 48 (r==0, lanes 0-15) is real
    if (rowoff == 0)
      obc[48 * DIM] = acc[3][0] + bp;
  }
}

extern "C" void kernel_launch(void* const* d_in, const int* in_sizes, int n_in,
                              void* d_out, int out_size, void* d_ws, size_t ws_size,
                              hipStream_t stream) {
  const float* x          = (const float*)d_in[0];
  const float* w_qkv      = (const float*)d_in[1];
  const float* w_proj     = (const float*)d_in[2];
  const float* b_proj     = (const float*)d_in[3];
  const float* bias_table = (const float*)d_in[4];
  const int*   rel_idx    = (const int*)d_in[5];

  char* ws = (char*)d_ws;
  __bf16* wq       = (__bf16*)ws;                                  // 884,736 B
  __bf16* wp       = (__bf16*)(ws + (size_t)JQ * 2);               // 294,912 B
  float*  bias_pad = (float*)(ws + (size_t)JQ * 2 + (size_t)JP * 2); // 196,608 B

  int B = in_sizes[0] / (NTOK * DIM);
  int total = JQ + JP + JB;
  pack_params<<<(total + 255) / 256, 256, 0, stream>>>(
      w_qkv, w_proj, bias_table, rel_idx, wq, wp, bias_pad);
  window_attn<<<B, 384, 0, stream>>>(x, wq, wp, bias_pad, b_proj, (float*)d_out);
}
